// PathCon_39041252720860
// MI455X (gfx1250) — matrix operand order repaired and checked by reading, not compiled, not run on hardware
//
#include <hip/hip_runtime.h>

typedef __attribute__((ext_vector_type(16))) __bf16 v16bf;
typedef __attribute__((ext_vector_type(8)))  __bf16 v8bf;
typedef __attribute__((ext_vector_type(8)))  float  v8f;

#define INPUT_DIM  256
#define NODE_DIM   6
#define OUT_DIM    256
#define NODE_REP   (INPUT_DIM + NODE_DIM)      // 262
#define K_REAL     (3*INPUT_DIM + 2*NODE_DIM)  // 780
#define K_PAD      800
#define K_TILES    (K_PAD/32)                  // 25
#define N_TILES    (OUT_DIM/16)                // 16
#define M_BLK      32
#define LDS_STRIDE 816                         // padded row stride (ushorts)

__device__ __forceinline__ unsigned short f2bf(float f) {
  union { float f; unsigned u; } x; x.f = f;
  unsigned u = x.u;
  unsigned r = u + 0x7FFFu + ((u >> 16) & 1u);  // round-to-nearest-even
  return (unsigned short)(r >> 16);
}

__device__ __forceinline__ unsigned f2bf_pair(float a, float b) {
  return (unsigned)f2bf(a) | ((unsigned)f2bf(b) << 16);
}

__device__ __forceinline__ v16bf cat16(v8bf lo, v8bf hi) {
  return __builtin_shufflevector(lo, hi, 0, 1, 2, 3, 4, 5, 6, 7,
                                         8, 9, 10, 11, 12, 13, 14, 15);
}

// ---------------------------------------------------------------- zero init
__global__ void pc_zero_kernel(float* __restrict__ nodeOut,
                               float* __restrict__ den, int nNode) {
  int total = nNode * NODE_REP + nNode;
  for (int i = blockIdx.x * blockDim.x + threadIdx.x; i < total;
       i += gridDim.x * blockDim.x) {
    if (i < nNode * NODE_REP) nodeOut[i] = 0.f;
    else                      den[i - nNode * NODE_REP] = 0.f;
  }
}

// ------------------------------------------- W -> bf16 WMMA B-fragment swizzle
// Fragment (kt,nt): per lane 16 contiguous bf16. lane&15 = N, lane>>4 = K-half.
__global__ __launch_bounds__(32)
void pc_wprep_kernel(const float* __restrict__ W, unsigned short* __restrict__ wfrag) {
  const int frag = blockIdx.x;              // kt*N_TILES + nt
  const int kt = frag / N_TILES;
  const int nt = frag % N_TILES;
  const int lane = threadIdx.x;
  const int n  = nt * 16 + (lane & 15);
  const int kb = kt * 32 + (lane >> 4) * 16;
  unsigned short* dst = wfrag + ((size_t)frag * 32 + lane) * 16;
  #pragma unroll
  for (int i = 0; i < 16; ++i) {
    int k = kb + i;
    float v = (k < K_REAL) ? W[(size_t)n * K_REAL + k] : 0.f;
    dst[i] = f2bf(v);
  }
}

// ---------------------------------------------------------------- scatter sum
__global__ __launch_bounds__(256)
void pc_scatter_kernel(const float* __restrict__ edgeAttr,
                       const float* __restrict__ mask,
                       const int*   __restrict__ col,
                       float* __restrict__ nodeOut,
                       float* __restrict__ den, int E) {
  const int e = blockIdx.x;
  if (e >= E) return;
  const int t = threadIdx.x;                // 0..255 == feature dim
  const int c = col[e];
  const float m = mask[e];
  atomicAdd(&nodeOut[(size_t)c * NODE_REP + t],
            edgeAttr[(size_t)e * INPUT_DIM + t] * m);
  if (t == 0) atomicAdd(&den[c], m);
}

// ------------------------------------------------- node_rep = num/(den+1) || x
__global__ __launch_bounds__(256)
void pc_finalize_kernel(const float* __restrict__ x,
                        const float* __restrict__ den,
                        float* __restrict__ nodeOut, int nNode) {
  const int n = blockIdx.x;
  if (n >= nNode) return;
  const int t = threadIdx.x;
  const float inv = 1.0f / (den[n] + 1.0f);
  if (t < INPUT_DIM) {
    size_t idx = (size_t)n * NODE_REP + t;
    nodeOut[idx] = nodeOut[idx] * inv;
  }
  if (t < NODE_DIM)
    nodeOut[(size_t)n * NODE_REP + INPUT_DIM + t] = x[(size_t)n * NODE_DIM + t];
}

// ---------------------------------------------------------------- WMMA GEMM
// edge_rep[E x 780(pad 800)] @ W^T -> edge_out[E x 256]
// block: 32 edges, 8 waves = 2(M) x 4(N); wave tile 16x64 -> 4 accumulators.
__global__ __launch_bounds__(256)
void pc_gemm_kernel(const float* __restrict__ nodeRep,   // [nNode][262] in d_out
                    const int*   __restrict__ rowIdx,
                    const int*   __restrict__ colIdx,
                    const float* __restrict__ edgeAttr,
                    const unsigned short* __restrict__ wfrag,
                    const float* __restrict__ bias,
                    float* __restrict__ edgeOut, int E) {
  __shared__ unsigned short sA[M_BLK * LDS_STRIDE];  // 52224 B
  __shared__ int sRow[M_BLK], sCol[M_BLK];

  const int tid   = threadIdx.x;
  const int eBase = blockIdx.x * M_BLK;
  const bool fullTile = (eBase + M_BLK <= E);

  if (tid < M_BLK) {
    int e = eBase + tid;
    sRow[tid] = (e < E) ? rowIdx[e] : 0;
    sCol[tid] = (e < E) ? colIdx[e] : 0;
  }
  __syncthreads();

  // Stage 32 bf16 edge_rep rows: [node_rep[row] | node_rep[col] | edge_attr | 0pad]
  // Segment boundaries (262/524/780) are even, so k-pairs never straddle:
  // convert 2 floats, pack, one ds_store_b32.
  {
    const int el = tid >> 3;     // 8 threads per edge row
    const int e  = eBase + el;
    const bool live = fullTile || (e < E);
    const int r = sRow[el], c = sCol[el];
    unsigned* dst = reinterpret_cast<unsigned*>(&sA[el * LDS_STRIDE]);
    for (int p = (tid & 7); p < K_PAD / 2; p += 8) {
      const int k = 2 * p;
      float a = 0.f, b = 0.f;
      if (live) {
        if (k < NODE_REP) {
          const float* s = &nodeRep[(size_t)r * NODE_REP + k];
          a = s[0]; b = s[1];
        } else if (k < 2 * NODE_REP) {
          const float* s = &nodeRep[(size_t)c * NODE_REP + (k - NODE_REP)];
          a = s[0]; b = s[1];
        } else if (k < K_REAL) {
          const float* s = &edgeAttr[(size_t)e * INPUT_DIM + (k - 2 * NODE_REP)];
          a = s[0]; b = s[1];
        }
      }
      dst[p] = f2bf_pair(a, b);
    }
  }
  __syncthreads();

  const int lane = tid & 31;
  const int wave = tid >> 5;
  const int wm  = wave & 1;      // which 16-edge half
  const int wn  = wave >> 1;     // which 64-output quarter
  const int l15 = lane & 15;
  const int hi  = lane >> 4;

  v8f acc[4] = {};
  const unsigned short* aRow = &sA[(wm * 16 + l15) * LDS_STRIDE];
  const unsigned short* wBase =
      wfrag + (size_t)((wn * 4) * 32 + lane) * 16;   // fragment nt = wn*4 + j

  for (int kt = 0; kt < K_TILES; ++kt) {
    // A fragment: contiguous K order (K = hi*16 + 0..15) -> two aligned 16B LDS reads
    const v8bf* p = reinterpret_cast<const v8bf*>(aRow + kt * 32 + hi * 16);
    const v16bf a = cat16(p[0], p[1]);   // K = hi*16+0..7 | hi*16+8..15

    // Load all four B fragments before any WMMA so the four matrix ops issue
    // back-to-back without accumulator register rotation.
    const unsigned short* wk = wBase + (size_t)kt * (N_TILES * 32 * 16);
    v16bf bm[4];
    #pragma unroll
    for (int j = 0; j < 4; ++j) {
      const v8bf* q = reinterpret_cast<const v8bf*>(wk + (size_t)j * (32 * 16));
      bm[j] = cat16(q[0], q[1]);
    }
    #pragma unroll
    for (int j = 0; j < 4; ++j) {
      acc[j] = __builtin_amdgcn_wmma_f32_16x16x32_bf16(
          false, a, false, bm[j], (short)0, acc[j], false, false);
    }
  }

  // C/D layout: VGPR r -> M = r + 8*hi, N = l15.
  // Fast path (tile fully in range): no per-store exec masking.
  if (fullTile) {
    #pragma unroll
    for (int j = 0; j < 4; ++j) {
      const int n  = wn * 64 + j * 16 + l15;
      const float bv = bias[n];
      float* outCol = edgeOut + (size_t)(eBase + wm * 16 + 8 * hi) * OUT_DIM + n;
      #pragma unroll
      for (int r = 0; r < 8; ++r) {
        outCol[(size_t)r * OUT_DIM] = acc[j][r] + bv;
      }
    }
  } else {
    #pragma unroll
    for (int j = 0; j < 4; ++j) {
      const int n  = wn * 64 + j * 16 + l15;
      const float bv = bias[n];
      #pragma unroll
      for (int r = 0; r < 8; ++r) {
        const int e = eBase + wm * 16 + 8 * hi + r;
        if (e < E) edgeOut[(size_t)e * OUT_DIM + n] = acc[j][r] + bv;
      }
    }
  }
}

// ---------------------------------------------------------------- launcher
extern "C" void kernel_launch(void* const* d_in, const int* in_sizes, int n_in,
                              void* d_out, int out_size, void* d_ws, size_t ws_size,
                              hipStream_t stream) {
  const float* x        = (const float*)d_in[0];
  const int*   eidx     = (const int*)d_in[2];
  const float* edgeAttr = (const float*)d_in[3];
  const float* mask     = (const float*)d_in[4];
  const float* W        = (const float*)d_in[5];
  const float* bias     = (const float*)d_in[6];

  const int nNode = in_sizes[0] / NODE_DIM;   // 10000
  const int E     = in_sizes[4];              // 320000
  const int* rowIdx = eidx;
  const int* colIdx = eidx + E;

  float* nodeOut = (float*)d_out;                          // [nNode][262]
  float* edgeOut = nodeOut + (size_t)nNode * NODE_REP;     // [E][256]

  // workspace: den (nNode f32) | wfrag (25*16 fragments * 512 bf16 = 400 KB)
  float* den = (float*)d_ws;
  size_t denBytes = ((size_t)nNode * sizeof(float) + 255) & ~(size_t)255;
  unsigned short* wfrag = (unsigned short*)((char*)d_ws + denBytes);

  const int zeroTotal = nNode * NODE_REP + nNode;
  hipLaunchKernelGGL(pc_zero_kernel, dim3((zeroTotal + 255) / 256), dim3(256),
                     0, stream, nodeOut, den, nNode);
  hipLaunchKernelGGL(pc_wprep_kernel, dim3(K_TILES * N_TILES), dim3(32),
                     0, stream, W, wfrag);
  hipLaunchKernelGGL(pc_scatter_kernel, dim3(E), dim3(256),
                     0, stream, edgeAttr, mask, colIdx, nodeOut, den, E);
  hipLaunchKernelGGL(pc_finalize_kernel, dim3(nNode), dim3(256),
                     0, stream, x, den, nodeOut, nNode);
  hipLaunchKernelGGL(pc_gemm_kernel, dim3((E + M_BLK - 1) / M_BLK), dim3(256),
                     0, stream, nodeOut, rowIdx, colIdx, edgeAttr, wfrag, bias,
                     edgeOut, E);
}
